// KnnMemory_13511967113708
// MI455X (gfx1250) — matrix operand order repaired and checked by reading, not compiled, not run on hardware
//
#include <hip/hip_runtime.h>
#include <hip/hip_bf16.h>

#define B_    4
#define N_    2048
#define DIM_  1024
#define K_    16384
#define TOPK_ 32
#define ROWS  (B_ * N_)     // 8192
#define MT    16            // rows per block
#define NWAVES 8
#define NT    256

typedef __attribute__((ext_vector_type(16))) __bf16 v16bf;
typedef __attribute__((ext_vector_type(16))) short  v16s;
typedef __attribute__((ext_vector_type(8)))  float  v8f;
typedef __attribute__((ext_vector_type(4)))  float  v4f;

union Frag { v16s s; v16bf bf; };

__device__ __forceinline__ unsigned short f2bf(float f) {
    unsigned int u = __float_as_uint(f);
    u += 0x7FFFu + ((u >> 16) & 1u);          // round-to-nearest-even
    return (unsigned short)(u >> 16);
}

// ---------------------------------------------------------------------------
// Kernel 0: convert queue (fp32 [DIM, K]) to bf16, pre-permuted into the
// WMMA B-fragment layout: qp[t][c][lane][e], t = 16-col tile, c = 32-deep
// K-chunk. B fragment load then = one contiguous 32B read per lane.
// B layout (16-bit 32x16): lanes 0-15 hold K=0..7,16..23 (2/VGPR), lanes 16-31
// hold K=8..15,24..31; n = lane%16.
// ---------------------------------------------------------------------------
__global__ __launch_bounds__(256) void pack_queue_bf16(
    const float* __restrict__ q, unsigned short* __restrict__ qp)
{
    unsigned int id = blockIdx.x * 256u + threadIdx.x;   // 0 .. 16M-1
    unsigned int e    = id & 15u;
    unsigned int l    = (id >> 4) & 31u;
    unsigned int c    = (id >> 9) & 31u;
    unsigned int t    = id >> 14;                        // 0..1023
    unsigned int half = l >> 4;
    unsigned int n    = l & 15u;
    unsigned int kk   = e + 8u * half + (e >= 8u ? 8u : 0u); // K within 32-chunk
    unsigned int d    = 32u * c + kk;                    // 0..1023
    unsigned int col  = t * 16u + n;                     // 0..16383
    qp[id] = f2bf(q[(size_t)d * K_ + col]);
}

// ---------------------------------------------------------------------------
// Kernel 1: fused bf16-WMMA GEMM (sim tile) + running top-32 per row.
// Block = 16 rows, 8 waves; wave w sweeps 32-col super-tiles w, w+8, ...
// (2 independent accumulators per wave -> no serial D->C WMMA chain,
//  A fragment reused across both tiles).
// ---------------------------------------------------------------------------
__global__ __launch_bounds__(NT) void gemm_topk(
    const float* __restrict__ x, const unsigned short* __restrict__ qp,
    float* __restrict__ wsVals, int* __restrict__ wsInds)
{
    __shared__ __align__(32) unsigned short ldsA[MT * DIM_]; // 32KB, A-frag layout
    __shared__ __align__(16) float simStage[NWAVES * 512];   // 16KB: [wave][row][32]

    const int tid  = threadIdx.x;
    const int wave = tid >> 5;
    const int lane = tid & 31;
    const int m0   = blockIdx.x * MT;

    // ---- Phase A: x tile -> bf16 -> LDS in A-fragment layout -------------
    // A layout (16-bit 16x32): lanes 0-15: M=lane, V0..3 K=0..7, V4..7 K=16..23
    //                          lanes16-31: same M, K ranges +8.
    for (int i = 0; i < 64; ++i) {
        int f = tid * 64 + i;
        int e = f & 15, l = (f >> 4) & 31, c = f >> 9;
        int m = l & 15, half = l >> 4;
        int v = e >> 1, bit = e & 1;
        int kk = (v < 4 ? 2 * v + bit : 16 + 2 * (v - 4) + bit) + 8 * half;
        int d  = 32 * c + kk;
        ldsA[f] = f2bf(x[(size_t)(m0 + m) * DIM_ + d]);
    }
    __syncthreads();

    // ---- running top-32 in registers (lanes 0..15 own one row each) ------
    float tv[TOPK_]; int ti[TOPK_];
    #pragma unroll
    for (int q = 0; q < TOPK_; ++q) { tv[q] = -__builtin_inff(); ti[q] = 0; }
    float minVal = -__builtin_inff();
    const int row  = lane & 15;
    const int half = lane >> 4, nn = lane & 15;

    // ---- Phase B: main loop over 32-wide column super-tiles --------------
    for (int tp = wave; tp < K_ / 32; tp += NWAVES) {
        v8f acc0 = {0.f, 0.f, 0.f, 0.f, 0.f, 0.f, 0.f, 0.f};
        v8f acc1 = {0.f, 0.f, 0.f, 0.f, 0.f, 0.f, 0.f, 0.f};
        const unsigned short* bp0 =
            qp + (size_t)(2 * tp) * 32u * 512u + (size_t)lane * 16u;
        const unsigned short* bp1 = bp0 + 32u * 512u;
        #pragma unroll 4
        for (int c = 0; c < 32; ++c) {
            Frag a, b0, b1;
            a.s  = *(const v16s*)&ldsA[(c * 32 + lane) * 16];
            b0.s = *(const v16s*)(bp0 + (size_t)c * 512u);
            b1.s = *(const v16s*)(bp1 + (size_t)c * 512u);
            __builtin_prefetch(bp0 + (size_t)(c + 1) * 512u, 0, 3);
            __builtin_prefetch(bp1 + (size_t)(c + 1) * 512u, 0, 3);
            acc0 = __builtin_amdgcn_wmma_f32_16x16x32_bf16(
                false, a.bf, false, b0.bf, (short)0, acc0, false, false);
            acc1 = __builtin_amdgcn_wmma_f32_16x16x32_bf16(
                false, a.bf, false, b1.bf, (short)0, acc1, false, false);
        }

        // stage both C tiles row-contiguously:
        // acc[r] of lane l = sim[r + 8*half][nn] ; row's 32 cands contiguous
        #pragma unroll
        for (int r = 0; r < 8; ++r) {
            simStage[wave * 512 + (r + 8 * half) * 32 + nn]      = acc0[r];
            simStage[wave * 512 + (r + 8 * half) * 32 + 16 + nn] = acc1[r];
        }

        // selection: lanes 0..15 scan their row's 32 new candidates, 4/ldsld
        if (lane < 16) {
            const float* sbase = &simStage[wave * 512 + row * 32];
            #pragma unroll 1
            for (int j0 = 0; j0 < 32; j0 += 4) {
                v4f vv = *(const v4f*)(sbase + j0);
                #pragma unroll
                for (int u = 0; u < 4; ++u) {
                    float val = vv[u];
                    int gidx  = tp * 32 + j0 + u;
                    if (val > minVal) {
                        int mp = 0; float mv = tv[0];
                        #pragma unroll
                        for (int q = 1; q < TOPK_; ++q)
                            if (tv[q] < mv) { mv = tv[q]; mp = q; }
                        #pragma unroll
                        for (int q = 0; q < TOPK_; ++q)
                            if (q == mp) { tv[q] = val; ti[q] = gidx; }
                        mv = tv[0];
                        #pragma unroll
                        for (int q = 1; q < TOPK_; ++q) mv = fminf(mv, tv[q]);
                        minVal = mv;
                    }
                }
            }
        }
    }
    __syncthreads();   // ldsA (A tile) dead from here; reuse for merge lists

    float* lv = (float*)ldsA;           // [NWAVES][MT][TOPK]  (16KB)
    int*   li = (int*)(ldsA + 8192);    // [NWAVES][MT][TOPK]  (16KB)
    if (lane < 16) {
        #pragma unroll
        for (int q = 0; q < TOPK_; ++q) {
            lv[(wave * MT + row) * TOPK_ + q] = tv[q];
            li[(wave * MT + row) * TOPK_ + q] = ti[q];
        }
    }
    __syncthreads();

    // merge 8x32 candidates per row -> sorted (descending) top-32
    if (wave == 0 && lane < 16) {
        for (int s = 0; s < TOPK_; ++s) {
            float best = -__builtin_inff(); int bp2 = 0;
            for (int p = 0; p < NWAVES * TOPK_; ++p) {
                float vv = lv[((p >> 5) * MT + row) * TOPK_ + (p & 31)];
                if (vv > best) { best = vv; bp2 = p; }
            }
            int w = bp2 >> 5, q = bp2 & 31;
            int bi = li[(w * MT + row) * TOPK_ + q];
            lv[(w * MT + row) * TOPK_ + q] = -__builtin_inff();
            wsVals[(size_t)(m0 + row) * TOPK_ + s] = best;
            wsInds[(size_t)(m0 + row) * TOPK_ + s] = bi;
        }
    }
}

// ---------------------------------------------------------------------------
// Kernel 2: softmax over top-32 + gather queue columns + weighted sum (fp32).
// ---------------------------------------------------------------------------
__global__ __launch_bounds__(256) void softmax_gather(
    const float* __restrict__ queue, const float* __restrict__ wsVals,
    const int* __restrict__ wsInds, float* __restrict__ out)
{
    __shared__ float w[TOPK_];
    __shared__ int   id[TOPK_];
    const int row = blockIdx.x;
    const int tid = threadIdx.x;

    if (tid < TOPK_) {                       // wave 0, all 32 lanes
        float v = wsVals[(size_t)row * TOPK_ + tid];
        int   i = wsInds[(size_t)row * TOPK_ + tid];
        float m = v;
        #pragma unroll
        for (int off = 16; off > 0; off >>= 1) m = fmaxf(m, __shfl_xor(m, off, 32));
        float e = __expf(v - m);
        float s = e;
        #pragma unroll
        for (int off = 16; off > 0; off >>= 1) s += __shfl_xor(s, off, 32);
        w[tid]  = e / s;
        id[tid] = i;
    }
    __syncthreads();

    for (int d = tid; d < DIM_; d += 256) {
        float acc = 0.f;
        #pragma unroll 8
        for (int j = 0; j < TOPK_; ++j)
            acc += w[j] * queue[(size_t)d * K_ + id[j]];
        out[(size_t)row * DIM_ + d] = acc;
    }
    if (tid < TOPK_) {
        int* oi = (int*)(out + (size_t)ROWS * DIM_);
        oi[(size_t)row * TOPK_ + tid] = id[tid];
    }
}

// ---------------------------------------------------------------------------
extern "C" void kernel_launch(void* const* d_in, const int* in_sizes, int n_in,
                              void* d_out, int out_size, void* d_ws, size_t ws_size,
                              hipStream_t stream)
{
    (void)in_sizes; (void)n_in; (void)out_size; (void)ws_size;
    const float* x     = (const float*)d_in[0];
    const float* queue = (const float*)d_in[1];
    float* out = (float*)d_out;

    // workspace: [0, 32MB) packed bf16 queue | [32MB,+1MB) vals | [+1MB) inds
    unsigned short* qp = (unsigned short*)d_ws;
    float* wsVals = (float*)((char*)d_ws + (size_t)DIM_ * K_ * 2);
    int*   wsInds = (int*)((char*)wsVals + (size_t)ROWS * TOPK_ * sizeof(float));

    pack_queue_bf16<<<(DIM_ * K_) / 256, 256, 0, stream>>>(queue, qp);
    gemm_topk<<<ROWS / MT, NT, 0, stream>>>(x, qp, wsVals, wsInds);
    softmax_gather<<<ROWS, 256, 0, stream>>>(queue, wsVals, wsInds, out);
}
